// Plugin_module_738734375274
// MI455X (gfx1250) — matrix operand order repaired and checked by reading, not compiled, not run on hardware
//
#include <hip/hip_runtime.h>
#include <math.h>

typedef __attribute__((ext_vector_type(16))) _Float16 v16h;
typedef __attribute__((ext_vector_type(8)))  _Float16 v8h;
typedef __attribute__((ext_vector_type(8)))  float    v8f;

#define HW (512*512)
#define EPSF 1e-12f

// fast reciprocal: single v_rcp_f32 (no IEEE div expansion)
__device__ __forceinline__ float frcp(float x) { return __builtin_amdgcn_rcpf(x); }

// ws layout (floats):
// [0,256)      stats:  B x 16 (9 used: SY,SY2,SL1,SL2,SG,SE,SLE1,SLE2,SPB)
// [256,4352)   cells:  B x 256 (16x16 grid sums of E)
// [4352,4608)  params: B x 16 (gamma[5], beta[5], alpha[5])
#define WS_CELLS 256
#define WS_PARAM 4352
#define WS_TOTAL 4608

// tap -> position-offset tables (clamped; padded taps hit ZERO rows of A)
__device__ __forceinline__ constexpr int dt1(int t) {   // conv1: 20-wide tile
    int tc = (t > 8) ? 8 : t;
    return (tc / 3) * 20 + (tc % 3);
}
__device__ __forceinline__ constexpr int dt2(int t) {   // conv2: 18-wide tile
    int tc = (t > 8) ? 8 : t;
    return (tc / 3) * 18 + (tc % 3);
}

__global__ __launch_bounds__(256) void k_zero(float* ws) {
    int i = blockIdx.x * 256 + threadIdx.x;
    if (i < WS_TOTAL) ws[i] = 0.f;
}

// ---------------- metrics pass: one block per (8-row band, batch) ----------------
__global__ __launch_bounds__(256) void k_stats(const float* __restrict__ ev,
                                               const float* __restrict__ fr,
                                               float* __restrict__ ws) {
    const int band = blockIdx.x;   // 0..63
    const int b    = blockIdx.y;   // 0..15
    const int tid  = threadIdx.x;
    const int y0   = band * 8;

    __shared__ float Yt[10 * 512];
    __shared__ float Et[10 * 512];
    __shared__ float acc[9];
    __shared__ float cell[16];
    if (tid < 9)  acc[tid]  = 0.f;
    if (tid < 16) cell[tid] = 0.f;

    // stage Y and E for rows y0-1 .. y0+8 (zero padded outside image)
    for (int idx = tid; idx < 10 * 512; idx += 256) {
        int yy = idx >> 9, x = idx & 511, gy = y0 - 1 + yy;
        float Y = 0.f, E = 0.f;
        if (gy >= 0 && gy < 512) {
            int o = gy * 512 + x;
            float R  = fr[(b*3+0)*HW + o];
            float G  = fr[(b*3+1)*HW + o];
            float Bc = fr[(b*3+2)*HW + o];
            Y = 0.299f*R + 0.587f*G + 0.114f*Bc;
            Y = fminf(fmaxf(Y, 0.f), 1.f);
            float P  = ev[(b*2+0)*HW + o];
            float N1 = ev[(b*2+1)*HW + o];
            E = fmaxf((P + N1) * (1.f/255.f), 0.f);
        }
        Yt[idx] = Y; Et[idx] = E;
    }
    __syncthreads();

    float sy=0.f, sy2=0.f, sl1=0.f, sl2=0.f, sg=0.f, se=0.f, sle1=0.f, sle2=0.f, spb=0.f;
    float cA = 0.f, cB = 0.f;

    for (int k = 0; k < 16; k++) {
        int idx = tid + k * 256;
        int yl = idx >> 9, x = idx & 511;
        int r = (yl + 1) * 512;
        float ymm = (x>0)   ? Yt[r-512 + x-1] : 0.f;
        float ym0 =           Yt[r-512 + x];
        float ymp = (x<511) ? Yt[r-512 + x+1] : 0.f;
        float y0m = (x>0)   ? Yt[r     + x-1] : 0.f;
        float yc  =           Yt[r     + x];
        float y0p = (x<511) ? Yt[r     + x+1] : 0.f;
        float ypm = (x>0)   ? Yt[r+512 + x-1] : 0.f;
        float yp0 =           Yt[r+512 + x];
        float ypp = (x<511) ? Yt[r+512 + x+1] : 0.f;

        float lap = ym0 + yp0 + y0m + y0p - 4.f*yc;
        float gx  = -ymm + ymp - 2.f*y0m + 2.f*y0p - ypm + ypp;
        float gy  = -ymm - 2.f*ym0 - ymp + ypm + 2.f*yp0 + ypp;
        float gm  = sqrtf(gx*gx + gy*gy);

        float em0 =           Et[r-512 + x];
        float e0m = (x>0)   ? Et[r     + x-1] : 0.f;
        float ec  =           Et[r     + x];
        float e0p = (x<511) ? Et[r     + x+1] : 0.f;
        float ep0 =           Et[r+512 + x];
        float lapE = em0 + ep0 + e0m + e0p - 4.f*ec;

        int go = (y0 + yl) * 512 + x;
        float P  = ev[(b*2+0)*HW + go] * (1.f/255.f);
        float N1 = ev[(b*2+1)*HW + go] * (1.f/255.f);
        float pb = (P - N1) * frcp(P + N1 + EPSF);

        sy += yc; sy2 += yc*yc; sl1 += lap; sl2 += lap*lap; sg += gm;
        se += ec; sle1 += lapE; sle2 += lapE*lapE; spb += pb;
        if (k & 1) cB += ec; else cA += ec;  // k even -> x=tid, k odd -> x=tid+256
    }

    atomicAdd(&acc[0], sy);  atomicAdd(&acc[1], sy2);
    atomicAdd(&acc[2], sl1); atomicAdd(&acc[3], sl2);
    atomicAdd(&acc[4], sg);  atomicAdd(&acc[5], se);
    atomicAdd(&acc[6], sle1);atomicAdd(&acc[7], sle2);
    atomicAdd(&acc[8], spb);
    atomicAdd(&cell[tid >> 5], cA);
    atomicAdd(&cell[(tid + 256) >> 5], cB);
    __syncthreads();

    if (tid < 9)  atomicAdd(&ws[b*16 + tid], acc[tid]);
    if (tid < 16) {
        int gyc = band >> 2;
        atomicAdd(&ws[WS_CELLS + b*256 + gyc*16 + tid], cell[tid]);
    }
}

// ---------------- tiny MLP head: one block per batch ----------------
__global__ __launch_bounds__(64) void k_mlp(const float* __restrict__ fe_w1, const float* __restrict__ fe_b1,
                                            const float* __restrict__ fe_w2, const float* __restrict__ fe_b2,
                                            const float* __restrict__ hg_w,  const float* __restrict__ hg_b,
                                            const float* __restrict__ hf_w,  const float* __restrict__ hf_b,
                                            float* __restrict__ ws) {
    const int b = blockIdx.x, tid = threadIdx.x;
    __shared__ float z[8];
    __shared__ float h1[64], h2[64];
    __shared__ float sb[64], sb2[64];
    __shared__ float gf[12];  // [0..9]=film, [10..11]=g

    float s = 0.f, s2 = 0.f;
    for (int j = 0; j < 4; j++) {
        float cm = ws[WS_CELLS + b*256 + tid*4 + j] * (1.f/1024.f);
        s += cm; s2 += cm*cm;
    }
    sb[tid] = s; sb2[tid] = s2;
    __syncthreads();

    if (tid == 0) {
        float S = 0.f, S2 = 0.f;
        for (int i = 0; i < 64; i++) { S += sb[i]; S2 += sb2[i]; }
        float mean = S * (1.f/256.f);
        float var  = S2 * (1.f/256.f) - mean*mean;
        float cv   = sqrtf(fmaxf(var, 0.f)) / (mean + EPSF);
        const float N = 262144.f;
        float SY  = ws[b*16+0], SY2 = ws[b*16+1], SL1 = ws[b*16+2], SL2 = ws[b*16+3];
        float SG  = ws[b*16+4], SE  = ws[b*16+5], SLE1= ws[b*16+6], SLE2= ws[b*16+7];
        float SPB = ws[b*16+8];
        float mu = SY / N;
        z[0] = mu;
        z[1] = sqrtf(fmaxf(SY2/N - mu*mu, 0.f));
        z[2] = (SL2 - SL1*SL1/N) / (N - 1.f);
        z[3] = SG / N;
        z[4] = cv;
        z[5] = SE / N;
        z[6] = SPB / N;
        z[7] = (SLE2 - SLE1*SLE1/N) / (N - 1.f);
    }
    __syncthreads();

    float a = fe_b1[tid];
    for (int k = 0; k < 8; k++) a += z[k] * fe_w1[tid*8 + k];
    h1[tid] = fmaxf(a, 0.f);
    __syncthreads();

    a = fe_b2[tid];
    for (int k = 0; k < 64; k++) a += h1[k] * fe_w2[tid*64 + k];
    h2[tid] = fmaxf(a, 0.f);
    __syncthreads();

    if (tid < 2) {
        float g = hg_b[tid];
        for (int k = 0; k < 64; k++) g += h2[k] * hg_w[tid*64 + k];
        gf[10 + tid] = frcp(1.f + __expf(-g));
    }
    if (tid < 10) {
        float f = hf_b[tid];
        for (int k = 0; k < 64; k++) f += h2[k] * hf_w[tid*64 + k];
        gf[tid] = f;
    }
    __syncthreads();

    if (tid < 5) {
        float gsum  = gf[10] + gf[11] + EPSF;
        float gamma = 1.f + 0.1f * gf[tid];
        float beta  = 0.1f * gf[5 + tid];
        float alpha = ((tid < 2) ? gf[10] : gf[11]) / gsum;
        ws[WS_PARAM + b*16 + tid]      = gamma;
        ws[WS_PARAM + b*16 + 5 + tid]  = beta;
        ws[WS_PARAM + b*16 + 10 + tid] = alpha;
    }
}

// ---------------- fused conv pipeline via WMMA: 16x16 tile per block ----------------
// Tap-major im2col K ordering:
//   conv1: K' = tap*8 + ch   (ch 0..4 real, 5..7 zero;  taps 9..11 zero) -> K=96, 3 WMMAs
//   conv2: K' = tap*16 + ch  (ch 0..15 real;            tap 9 zero)      -> K=160, 5 WMMAs
// A-fragments carry the zeros; B-fragments become contiguous LDS vector loads.
__global__ __launch_bounds__(256) void k_conv(const float* __restrict__ ev, const float* __restrict__ fr,
    const float* __restrict__ c1w,
    const float* __restrict__ bn1g, const float* __restrict__ bn1b,
    const float* __restrict__ bn1m, const float* __restrict__ bn1v,
    const float* __restrict__ c2w,
    const float* __restrict__ bn2g, const float* __restrict__ bn2b,
    const float* __restrict__ bn2m, const float* __restrict__ bn2v,
    const float* __restrict__ res, const float* __restrict__ ws,
    float* __restrict__ outA, float* __restrict__ outB) {
    const int bx = blockIdx.x, by = blockIdx.y, b = blockIdx.z;
    const int tid  = threadIdx.x;
    const int lane = tid & 31, wave = tid >> 5;
    const int hi   = lane >> 4;     // 0: lanes 0-15, 1: lanes 16-31
    const int mrow = lane & 15;

    __shared__ float sin_[5 * 400];     // 20x20 input tile, 5 channels, f32 (residual + reference)
    __shared__ v8h   sin_h[400];        // 20x20 tile as [pos][8ch] f16 (ch 5..7 zero)
    __shared__ v16h  sfx[324];          // 18x18 conv1 result as [pos][16ch] f16
    __shared__ float sbn1i[16], sbn1a[16], sbn2i[8], sbn2a[8];
    __shared__ float sgam[5], sbet[5], salp[5], sres[1];

    // warm L2 for this block's tile rows (global_prefetch_b8)
    if (tid < 20) {
        int gy = by*16 - 2 + tid;
        if (gy >= 0 && gy < 512) {
            int gx = bx*16 - 2; gx = (gx < 0) ? 0 : gx;
            __builtin_prefetch(&ev[(b*2+0)*HW + gy*512 + gx], 0, 0);
            __builtin_prefetch(&ev[(b*2+1)*HW + gy*512 + gx], 0, 0);
            __builtin_prefetch(&fr[(b*3+0)*HW + gy*512 + gx], 0, 0);
            __builtin_prefetch(&fr[(b*3+1)*HW + gy*512 + gx], 0, 0);
            __builtin_prefetch(&fr[(b*3+2)*HW + gy*512 + gx], 0, 0);
        }
    }

    // stage input tile (zero-padded SAME borders): f32 copy + f16 [pos][8] copy
    {
        const int gy0 = by*16 - 2, gx0 = bx*16 - 2;
        _Float16* sh = (_Float16*)sin_h;
        for (int idx = tid; idx < 2000; idx += 256) {
            int ci = idx / 400, rem = idx - ci*400, yy = rem / 20, xx = rem - yy*20;
            int gy = gy0 + yy, gx = gx0 + xx;
            float v = 0.f;
            if (gy >= 0 && gy < 512 && gx >= 0 && gx < 512) {
                int o = gy*512 + gx;
                v = (ci < 2) ? ev[(b*2+ci)*HW + o] : fr[(b*3+ci-2)*HW + o];
            }
            sin_[idx] = v;
            sh[rem*8 + ci] = (_Float16)v;
        }
        for (int idx = tid; idx < 1200; idx += 256) {   // zero pad channels 5..7
            int pos = idx / 3, j = idx - pos*3;
            sh[pos*8 + 5 + j] = (_Float16)0.f;
        }
    }
    if (tid < 16) {
        float inv = bn1g[tid] * rsqrtf(bn1v[tid] + 1e-5f);
        sbn1i[tid] = inv; sbn1a[tid] = bn1b[tid] - bn1m[tid]*inv;
    }
    if (tid < 5) {
        float inv = bn2g[tid] * rsqrtf(bn2v[tid] + 1e-5f);
        sbn2i[tid] = inv; sbn2a[tid] = bn2b[tid] - bn2m[tid]*inv;
        sgam[tid] = ws[WS_PARAM + b*16 + tid];
        sbet[tid] = ws[WS_PARAM + b*16 + 5 + tid];
        salp[tid] = ws[WS_PARAM + b*16 + 10 + tid];
    }
    if (tid == 0) sres[0] = res[0];
    __syncthreads();

    // weight A-fragments (contiguous 16-bit A layout: element i -> K = i + 16*hi).
    // Zeros in padded slots let the B side hold clamped (finite) garbage.
    v16h a1[3], a2[5];
    #pragma unroll
    for (int f = 0; f < 3; f++)
        #pragma unroll
        for (int i = 0; i < 16; i++) {
            int K  = f*32 + (i + (hi ? 16 : 0));
            int t  = K >> 3, ci = K & 7;                 // tap-major, 8-wide
            a1[f][i] = (t < 9 && ci < 5) ? (_Float16)c1w[mrow*45 + ci*9 + t] : (_Float16)0.f;
        }
    #pragma unroll
    for (int f = 0; f < 5; f++)
        #pragma unroll
        for (int i = 0; i < 16; i++) {
            int K  = f*32 + (i + (hi ? 16 : 0));
            int t  = K >> 4, ci = K & 15;                // tap-major, 16-wide
            a2[f][i] = (t < 9 && mrow < 5) ? (_Float16)c2w[mrow*144 + ci*9 + t] : (_Float16)0.f;
        }

    // ---- conv1 over 18x18 halo region (324 positions): 3 WMMAs, B = 2x v8h LDS loads ----
    for (int s = wave; s < 21; s += 8) {
        int p  = s*16 + mrow;
        int pc = (p < 324) ? p : 323;
        int r = pc / 18, c = pc - r*18;
        const int p0 = r*20 + c;
        v8f acc = {};
        #pragma unroll
        for (int f = 0; f < 3; f++) {
            // lane-half taps: lo -> 4f,4f+1 ; hi -> 4f+2,4f+3 (compile-time offsets)
            const int dA = hi ? dt1(4*f + 2) : dt1(4*f + 0);
            const int dB = hi ? dt1(4*f + 3) : dt1(4*f + 1);
            v8h lo = sin_h[p0 + dA];
            v8h hh = sin_h[p0 + dB];
            v16h bf = __builtin_shufflevector(lo, hh, 0,1,2,3,4,5,6,7,8,9,10,11,12,13,14,15);
            acc = __builtin_amdgcn_wmma_f32_16x16x32_f16(false, a1[f], false, bf,
                                                         (short)0, acc, false, false);
        }
        _Float16* sfp = (_Float16*)sfx;
        #pragma unroll
        for (int rr = 0; rr < 8; rr++) {
            int m = rr + (hi ? 8 : 0);
            float v = acc[rr]*sbn1i[m] + sbn1a[m];
            v = v * frcp(1.f + __expf(-v));              // SiLU (single v_rcp_f32)
            if (p < 324) sfp[p*16 + m] = (_Float16)v;
        }
    }
    __syncthreads();

    // ---- conv2 over 16x16 outputs: 5 WMMAs, B = ONE v16h LDS load per fragment ----
    const float rs = sres[0];
    for (int s = wave; s < 16; s += 8) {
        const int oy = s, ox = mrow;
        const int p0 = oy*18 + ox;
        v8f acc = {};
        #pragma unroll
        for (int f = 0; f < 5; f++) {
            // lane-half tap: lo -> 2f ; hi -> 2f+1 (tap 9 clamped; A holds zeros there)
            const int d = hi ? dt2(2*f + 1) : dt2(2*f + 0);
            v16h bf = sfx[p0 + d];
            acc = __builtin_amdgcn_wmma_f32_16x16x32_f16(false, a2[f], false, bf,
                                                         (short)0, acc, false, false);
        }
        const int gy = by*16 + oy, gx = bx*16 + ox;
        #pragma unroll
        for (int rr = 0; rr < 8; rr++) {
            int m = rr + (hi ? 8 : 0);
            if (m < 5) {
                float v = acc[rr]*sbn2i[m] + sbn2a[m];
                v = sgam[m]*v + sbet[m];
                v *= salp[m];
                float base = sin_[m*400 + (oy + 2)*20 + (ox + 2)];
                float o = base + rs*v;
                if (m < 2) outA[((b*2 + m)*512 + gy)*512 + gx] = o;
                else       outB[((b*3 + (m - 2))*512 + gy)*512 + gx] = o;
            }
        }
    }
}

extern "C" void kernel_launch(void* const* d_in, const int* in_sizes, int n_in,
                              void* d_out, int out_size, void* d_ws, size_t ws_size,
                              hipStream_t stream) {
    (void)in_sizes; (void)n_in; (void)out_size; (void)ws_size;
    const float* ev    = (const float*)d_in[0];
    const float* fr    = (const float*)d_in[1];
    const float* c1w   = (const float*)d_in[2];
    const float* bn1g  = (const float*)d_in[3];
    const float* bn1b  = (const float*)d_in[4];
    const float* bn1m  = (const float*)d_in[5];
    const float* bn1v  = (const float*)d_in[6];
    const float* c2w   = (const float*)d_in[7];
    const float* bn2g  = (const float*)d_in[8];
    const float* bn2b  = (const float*)d_in[9];
    const float* bn2m  = (const float*)d_in[10];
    const float* bn2v  = (const float*)d_in[11];
    const float* fe_w1 = (const float*)d_in[12];
    const float* fe_b1 = (const float*)d_in[13];
    const float* fe_w2 = (const float*)d_in[14];
    const float* fe_b2 = (const float*)d_in[15];
    const float* hg_w  = (const float*)d_in[16];
    const float* hg_b  = (const float*)d_in[17];
    const float* hf_w  = (const float*)d_in[18];
    const float* hf_b  = (const float*)d_in[19];
    const float* rsc   = (const float*)d_in[20];

    float* ws   = (float*)d_ws;
    float* outA = (float*)d_out;
    float* outB = outA + (size_t)16 * 2 * HW;   // second tuple element

    k_zero<<<(WS_TOTAL + 255)/256, 256, 0, stream>>>(ws);
    k_stats<<<dim3(64, 16), 256, 0, stream>>>(ev, fr, ws);
    k_mlp<<<16, 64, 0, stream>>>(fe_w1, fe_b1, fe_w2, fe_b2, hg_w, hg_b, hf_w, hf_b, ws);
    k_conv<<<dim3(32, 32, 16), 256, 0, stream>>>(ev, fr, c1w, bn1g, bn1b, bn1m, bn1v,
                                                 c2w, bn2g, bn2b, bn2m, bn2v,
                                                 rsc, ws, outA, outB);
}
